// TiteSelfAttention_78211354460284
// MI455X (gfx1250) — matrix operand order repaired and checked by reading, compiled, not run on hardware
//
#include <hip/hip_runtime.h>
#include <hip/hip_bf16.h>

// ---------------------------------------------------------------------------
// TiteSelfAttention for MI455X (gfx1250, wave32, WMMA, async global->LDS)
//   B=8, S=2048, HID=1024, NH=16, HD=64, pool K=2, SP=1024
// Pipeline:
//   0) cvt_bf16  : X, W -> bf16 once (keeps all conversion out of hot loops)
//   1) qkv_gemm  : 64x64 tile / 8 waves, BK=64, double-buffered
//                  GLOBAL_LOAD_ASYNC_TO_LDS_B128 bf16 staging, WMMA bf16,
//                  bias fused; Q/K stored (b,h,s,d), V stored (b,h,d,s).
//   2) pool_rope_q / rope_k : elementwise pool + RoPE
//   3) attn      : flash attention, 64-key tiles (16 WMMA per tile, halved
//                  per-key softmax VALU), online softmax via 16-lane shuffles
// ---------------------------------------------------------------------------

typedef __attribute__((ext_vector_type(16))) __bf16 bf16x16;
typedef __attribute__((ext_vector_type(8)))  __bf16 bf16x8;
typedef __attribute__((ext_vector_type(8)))  float  f32x8;
typedef __attribute__((ext_vector_type(4)))  int    i32x4;

#define NEG_BIG (-3.0e38f)

// ---- CDNA5 async global->LDS (ASYNCcnt) with guarded fallback -------------
#if defined(__HIP_DEVICE_COMPILE__) && __has_builtin(__builtin_amdgcn_global_load_async_to_lds_b128)
#define HAS_ASYNC 1
#else
#define HAS_ASYNC 0
#endif

__device__ __forceinline__ void cp16_to_lds(const void* g, void* l) {
#if HAS_ASYNC
  __builtin_amdgcn_global_load_async_to_lds_b128(
      (__attribute__((address_space(1))) i32x4*)g,
      (__attribute__((address_space(3))) i32x4*)l, 0, 0);
#else
  *(i32x4*)l = *(const i32x4*)g;
#endif
}

__device__ __forceinline__ void wait_async_lds() {
#if HAS_ASYNC
#if __has_builtin(__builtin_amdgcn_s_wait_asynccnt)
  __builtin_amdgcn_s_wait_asynccnt(0);
#else
  asm volatile("s_wait_asynccnt 0" ::: "memory");
#endif
#endif
}

// ---- WMMA operand loaders -------------------------------------------------
// A-operand (16x32 bf16): lane<16 -> M=lane, K=0..15 contiguous;
// lane>=16 -> M=lane-16, K=16..31. Two 16B-aligned 8-half runs.
__device__ __forceinline__ bf16x16 load_a16(const __bf16* p0, const __bf16* p1) {
  bf16x8 lo = *(const bf16x8*)p0;
  bf16x8 hi = *(const bf16x8*)p1;
  bf16x16 r;
#pragma unroll
  for (int i = 0; i < 8; ++i) { r[i] = lo[i]; r[i + 8] = hi[i]; }
  return r;
}
// B-operand (32x16 bf16): lane holds column N=lane&15, K=(lane>=16?16:0)+i:
// 16 contiguous halves; two 16B loads (LDS rows only 16B aligned).
__device__ __forceinline__ bf16x16 load_b16c(const __bf16* p) {
  return load_a16(p, p + 8);
}
__device__ __forceinline__ bf16x16 load_b16(const __bf16* p) {  // 32B aligned
  return *(const bf16x16*)p;
}

__device__ __forceinline__ float red_max16(float v) {
#pragma unroll
  for (int m = 8; m >= 1; m >>= 1) v = fmaxf(v, __shfl_xor(v, m, 32));
  return v;
}
__device__ __forceinline__ float red_sum16(float v) {
#pragma unroll
  for (int m = 8; m >= 1; m >>= 1) v += __shfl_xor(v, m, 32);
  return v;
}

// ---------------------------------------------------------------------------
// Kernel 0: f32 -> bf16 conversion (8 elements per thread, vectorized).
// ---------------------------------------------------------------------------
__global__ void cvt_bf16(const float* __restrict__ src, __bf16* __restrict__ dst) {
  const size_t i = ((size_t)blockIdx.x * blockDim.x + threadIdx.x) * 8;
  float4 a = *(const float4*)(src + i);
  float4 b = *(const float4*)(src + i + 4);
  bf16x8 o;
  o[0] = (__bf16)a.x; o[1] = (__bf16)a.y; o[2] = (__bf16)a.z; o[3] = (__bf16)a.w;
  o[4] = (__bf16)b.x; o[5] = (__bf16)b.y; o[6] = (__bf16)b.z; o[7] = (__bf16)b.w;
  *(bf16x8*)(dst + i) = o;
}

// ---------------------------------------------------------------------------
// Kernel 1: QKV projection. 256 threads = 8 waves; tile 64(M)x64(N), BK=64,
// double-buffered async bf16 LDS staging (4 WMMA per wave per buffer).
// ---------------------------------------------------------------------------
#define LDB 72  // bf16 row stride in LDS: 144B = 9x16B (aligned chunks, 16 banks)

__global__ void __launch_bounds__(256)
qkv_gemm(const __bf16* __restrict__ Xb, const __bf16* __restrict__ Wb,
         const float* __restrict__ bias,
         __bf16* __restrict__ qraw, __bf16* __restrict__ kbuf,
         __bf16* __restrict__ vT) {
  __shared__ __bf16 As[2][64 * LDB];
  __shared__ __bf16 Bs[2][64 * LDB];

  const int tid  = threadIdx.x;
  const int lane = tid & 31;
  const int wave = tid >> 5;
  const int ln = lane & 15, hi = lane >> 4;
  const int wm = wave & 3;   // 0..3 : 16-row sub-block
  const int wn = wave >> 2;  // 0..1 : 32-col sub-block

  const int n0 = blockIdx.x * 64;  // 0..3071
  const int m0 = blockIdx.y * 64;  // 0..16383

  // Stage one 64x64 bf16 tile of A and B: 512 16B chunks each, 2 per thread.
  auto issue = [&](int buf, int kk) {
#pragma unroll
    for (int j = 0; j < 2; ++j) {
      const int c = tid + j * 256;
      const int row = c >> 3, col8 = (c & 7) * 8;
      cp16_to_lds(Xb + (size_t)(m0 + row) * 1024 + kk + col8,
                  &As[buf][row * LDB + col8]);
      cp16_to_lds(Wb + (size_t)(n0 + row) * 1024 + kk + col8,
                  &Bs[buf][row * LDB + col8]);
    }
  };

  f32x8 acc[2] = {};

  issue(0, 0);
  wait_async_lds();
  __syncthreads();

  for (int i = 0; i < 16; ++i) {  // K = 1024 in BK=64 steps
    const int cur = i & 1;
    if (i + 1 < 16) issue(cur ^ 1, (i + 1) * 64);

#pragma unroll
    for (int kl = 0; kl < 64; kl += 32) {
      const __bf16* ab = &As[cur][(wm * 16 + ln) * LDB + kl];
      bf16x16 a = load_a16(ab + hi * 16, ab + hi * 16 + 8);
#pragma unroll
      for (int t = 0; t < 2; ++t) {
        const __bf16* bb = &Bs[cur][(wn * 32 + t * 16 + ln) * LDB + kl + hi * 16];
        acc[t] = __builtin_amdgcn_wmma_f32_16x16x32_bf16(false, a, false, load_b16c(bb),
                                                         (short)0, acc[t], false, false);
      }
    }

    wait_async_lds();   // next buffer's async loads complete
    __syncthreads();    // all waves done reading cur before overwrite
  }

  // Epilogue: bias + routed bf16 stores (each 16-col tile: one section/head).
  const int b  = m0 >> 11;
  const int sb = (m0 & 2047) + wm * 16;
#pragma unroll
  for (int t = 0; t < 2; ++t) {
    const int n  = n0 + wn * 32 + t * 16;
    const int sec = n >> 10;
    const int cc  = n & 1023;
    const int h   = cc >> 6;
    const int d   = (cc & 63) + ln;
    const float bv = bias[n + ln];
    const size_t bh = (size_t)b * 16 + h;
#pragma unroll
    for (int r = 0; r < 8; ++r) {
      const int s = sb + r + hi * 8;  // C layout: VGPR r, lane half -> row
      const float v = acc[t][r] + bv;
      if (sec == 0)      qraw[(bh * 2048 + s) * 64 + d] = (__bf16)v;
      else if (sec == 1) kbuf[(bh * 2048 + s) * 64 + d] = (__bf16)v;
      else               vT[(bh * 64 + d) * 2048 + s]   = (__bf16)v;
    }
  }
}

// ---------------------------------------------------------------------------
// Kernel 2a: masked 2:1 average pool of q + RoPE at pooled positions.
// ---------------------------------------------------------------------------
__global__ void pool_rope_q(const __bf16* __restrict__ qraw,
                            const unsigned char* __restrict__ am,
                            __bf16* __restrict__ q2) {
  const int idx = blockIdx.x * blockDim.x + threadIdx.x;  // < 8*16*1024*32
  const int j  = idx & 31;
  const int p  = (idx >> 5) & 1023;
  const int bh = idx >> 15;
  const int b  = bh >> 4;

  const __bf16* src = qraw + ((size_t)bh * 2048 + 2 * p) * 64 + 2 * j;
  const int m0 = am[b * 2048 + 2 * p] ? 1 : 0;
  const int m1 = am[b * 2048 + 2 * p + 1] ? 1 : 0;
  const float inv = 1.0f / (float)max(m0 + m1, 1);
  float x0 = ((float)src[0] * m0 + (float)src[64] * m1) * inv;
  float x1 = ((float)src[1] * m0 + (float)src[65] * m1) * inv;

  const float theta = __powf(10000.0f, -(float)(2 * j) * (1.0f / 64.0f));
  float sn, cs;
  sincosf((float)p * theta, &sn, &cs);

  __bf16* dst = q2 + ((size_t)bh * 1024 + p) * 64 + 2 * j;
  dst[0] = (__bf16)(x0 * cs - x1 * sn);
  dst[1] = (__bf16)(x1 * cs + x0 * sn);
}

// ---------------------------------------------------------------------------
// Kernel 2b: RoPE on k, in place.
// ---------------------------------------------------------------------------
__global__ void rope_k(__bf16* __restrict__ kbuf) {
  const int idx = blockIdx.x * blockDim.x + threadIdx.x;  // < 8*16*2048*32
  const int j  = idx & 31;
  const int s  = (idx >> 5) & 2047;
  const int bh = idx >> 16;

  __bf16* ptr = kbuf + ((size_t)bh * 2048 + s) * 64 + 2 * j;
  const float x0 = (float)ptr[0], x1 = (float)ptr[1];
  const float theta = __powf(10000.0f, -(float)(2 * j) * (1.0f / 64.0f));
  float sn, cs;
  sincosf((float)s * theta, &sn, &cs);
  ptr[0] = (__bf16)(x0 * cs - x1 * sn);
  ptr[1] = (__bf16)(x1 * cs + x0 * sn);
}

// ---------------------------------------------------------------------------
// Kernel 3: flash-style attention. One wave per (b, h, 16-query tile),
// 64-key tiles: 16 WMMA + one softmax update per tile.
// ---------------------------------------------------------------------------
#define PLD 72  // padded P-tile stride (144B = 9x16B, conflict-free reads)

__global__ void __launch_bounds__(32)
attn(const __bf16* __restrict__ q2, const __bf16* __restrict__ kbuf,
     const __bf16* __restrict__ vT, const unsigned char* __restrict__ am,
     float* __restrict__ out) {
  __shared__ __bf16 pl[16 * PLD];  // P tile staging: C-layout -> A-layout

  const int lane = threadIdx.x;
  const int ln = lane & 15, hi = lane >> 4;
  const int qt = blockIdx.x;  // 0..63
  const int h  = blockIdx.y;
  const int b  = blockIdx.z;
  const size_t bh = (size_t)b * 16 + h;

  const __bf16* qp = q2 + (bh * 1024 + (size_t)qt * 16) * 64;
  const __bf16* kp = kbuf + bh * 2048 * 64;
  const __bf16* vp = vT + bh * 64 * 2048;
  const unsigned char* mk = am + (size_t)b * 2048;

  // Q A-operands; fold 1/sqrt(64)=0.125 into A (exact exponent shift in bf16).
  bf16x16 aq0 = load_a16(qp + ln * 64 + hi * 16,      qp + ln * 64 + hi * 16 + 8);
  bf16x16 aq1 = load_a16(qp + ln * 64 + 32 + hi * 16, qp + ln * 64 + 40 + hi * 16);
#pragma unroll
  for (int i = 0; i < 16; ++i) {
    aq0[i] = (__bf16)((float)aq0[i] * 0.125f);
    aq1[i] = (__bf16)((float)aq1[i] * 0.125f);
  }

  f32x8 O[4] = {};
  float m[8], l[8];
#pragma unroll
  for (int r = 0; r < 8; ++r) { m[r] = NEG_BIG; l[r] = 0.0f; }

  for (int kc = 0; kc < 2048; kc += 64) {
    // Scores: four 16x16 C tiles covering 64 keys.
    f32x8 s[4];
    bool ok[4];
#pragma unroll
    for (int c = 0; c < 4; ++c) {
      const int kk = kc + c * 16 + ln;
      ok[c] = mk[kk] != 0;
      bf16x16 blo = load_b16(kp + (size_t)kk * 64 + hi * 16);
      bf16x16 bhi2 = load_b16(kp + (size_t)kk * 64 + 32 + hi * 16);
      f32x8 z = {};
      s[c] = __builtin_amdgcn_wmma_f32_16x16x32_bf16(false, aq0, false, blo,
                                                     (short)0, z, false, false);
      s[c] = __builtin_amdgcn_wmma_f32_16x16x32_bf16(false, aq1, false, bhi2,
                                                     (short)0, s[c], false, false);
    }

    // Online softmax update (one per 64 keys); write P tile to LDS in-loop.
#pragma unroll
    for (int r = 0; r < 8; ++r) {
      float v0 = ok[0] ? s[0][r] : NEG_BIG;
      float v1 = ok[1] ? s[1][r] : NEG_BIG;
      float v2 = ok[2] ? s[2][r] : NEG_BIG;
      float v3 = ok[3] ? s[3][r] : NEG_BIG;
      const float rm = red_max16(fmaxf(fmaxf(v0, v1), fmaxf(v2, v3)));
      const float nm = fmaxf(m[r], rm);
      const float f  = __expf(m[r] - nm);   // m==nm==NEG_BIG -> f=1, O is 0
      const float e0 = ok[0] ? __expf(v0 - nm) : 0.0f;
      const float e1 = ok[1] ? __expf(v1 - nm) : 0.0f;
      const float e2 = ok[2] ? __expf(v2 - nm) : 0.0f;
      const float e3 = ok[3] ? __expf(v3 - nm) : 0.0f;
      const float rs = red_sum16((e0 + e1) + (e2 + e3));
      m[r] = nm;
      l[r] = l[r] * f + rs;
      O[0][r] *= f; O[1][r] *= f; O[2][r] *= f; O[3][r] *= f;
      const int row = r + hi * 8;
      pl[row * PLD + ln]      = (__bf16)e0;
      pl[row * PLD + 16 + ln] = (__bf16)e1;
      pl[row * PLD + 32 + ln] = (__bf16)e2;
      pl[row * PLD + 48 + ln] = (__bf16)e3;
    }

    // P as two A-operands (keys [0,32) and [32,64)); single wave, DS in-order.
    const __bf16* pr = pl + ln * PLD;
    bf16x16 ap0 = load_a16(pr + hi * 16,      pr + hi * 16 + 8);
    bf16x16 ap1 = load_a16(pr + 32 + hi * 16, pr + 40 + hi * 16);

    // O += P x V : V^T layout gives contiguous-key B-operands.
#pragma unroll
    for (int t = 0; t < 4; ++t) {
      const __bf16* vb = vp + (size_t)(t * 16 + ln) * 2048 + kc + hi * 16;
      O[t] = __builtin_amdgcn_wmma_f32_16x16x32_bf16(false, ap0, false, load_b16(vb),
                                                     (short)0, O[t], false, false);
      O[t] = __builtin_amdgcn_wmma_f32_16x16x32_bf16(false, ap1, false, load_b16(vb + 32),
                                                     (short)0, O[t], false, false);
    }
  }

#pragma unroll
  for (int r = 0; r < 8; ++r) {
    const int row = r + hi * 8;
    const int p = qt * 16 + row;
    const float inv = l[r] > 0.0f ? 1.0f / l[r] : 0.0f;
    const float qm = (mk[2 * p] | mk[2 * p + 1]) ? 1.0f : 0.0f;
    const float sc = inv * qm;
    const size_t base = (((size_t)b * 1024 + p) * 16 + h) * 64;
    out[base + 0  + ln] = O[0][r] * sc;
    out[base + 16 + ln] = O[1][r] * sc;
    out[base + 32 + ln] = O[2][r] * sc;
    out[base + 48 + ln] = O[3][r] * sc;
  }
}

// ---------------------------------------------------------------------------
// Host launcher
// ---------------------------------------------------------------------------
extern "C" void kernel_launch(void* const* d_in, const int* in_sizes, int n_in,
                              void* d_out, int out_size, void* d_ws, size_t ws_size,
                              hipStream_t stream) {
  const float* X            = (const float*)d_in[0];          // (8,2048,1024)
  const unsigned char* am   = (const unsigned char*)d_in[1];  // (8,2048) bool
  const float* W            = (const float*)d_in[2];          // (3072,1024)
  const float* bias         = (const float*)d_in[3];          // (3072,)
  float* out                = (float*)d_out;                  // (8,1024,1024)

  // Workspace (bf16): qraw 32MB | k 32MB | v^T 32MB | q2 16MB | Xb 32MB | Wb 6MB
  __bf16* qraw = (__bf16*)d_ws;
  __bf16* kbuf = qraw + (size_t)16777216;
  __bf16* vT   = kbuf + (size_t)16777216;
  __bf16* q2   = vT   + (size_t)16777216;
  __bf16* Xb   = q2   + (size_t)8388608;
  __bf16* Wb   = Xb   + (size_t)16777216;

  cvt_bf16<<<dim3(8192), 256, 0, stream>>>(X, Xb);   // 16,777,216 elems
  cvt_bf16<<<dim3(1536), 256, 0, stream>>>(W, Wb);   //  3,145,728 elems
  qkv_gemm<<<dim3(48, 256), 256, 0, stream>>>(Xb, Wb, bias, qraw, kbuf, vT);
  pool_rope_q<<<dim3(16384), 256, 0, stream>>>(qraw, am, q2);
  rope_k<<<dim3(32768), 256, 0, stream>>>(kbuf);
  attn<<<dim3(64, 16, 8), 32, 0, stream>>>(q2, kbuf, vT, am, out);
}
